// BuySellLinkPredictionNoEmbedding_8418135900644
// MI455X (gfx1250) — compile-verified, matrix operand order checked by reading, not run on hardware
//
#include <hip/hip_runtime.h>
#include <cstdint>

// Problem constants (match reference)
#define NCN 10000
#define NTN 5000
#define DD  32
#define OO  32
#define FE  8
#define HH  64
#define EE  50000
#define LL  50000
#define KTOT 1056           // 32*32 outer-product rows + 32 bias rows
#define NTILES (EE / 16)    // 3125, exact
#define NPAIRS ((NTILES + 1) / 2)

typedef __attribute__((ext_vector_type(16))) __bf16 v16bf;
typedef __attribute__((ext_vector_type(8)))  float  v8f;

union BV { v16bf v; uint4 q[2]; };

#define ENC_NEGINF 0x007FFFFFu
__device__ __forceinline__ unsigned enc_ord(float f) {
  unsigned u = __float_as_uint(f);
  return (u & 0x80000000u) ? ~u : (u | 0x80000000u);
}
__device__ __forceinline__ float dec_ord(unsigned e) {
  unsigned u = (e & 0x80000000u) ? (e ^ 0x80000000u) : ~e;
  return __uint_as_float(u);
}

// out[n,o] = b[o] + sum_d x[n,d]*W[d,o]   (D=O=32)
__global__ void k_linear32(const float* __restrict__ x, const float* __restrict__ W,
                           const float* __restrict__ b, float* __restrict__ out, int N) {
  int t = blockIdx.x * blockDim.x + threadIdx.x;
  if (t >= N * 32) return;
  int n = t >> 5, o = t & 31;
  float acc = b[o];
  #pragma unroll 8
  for (int d = 0; d < 32; ++d) acc = fmaf(x[n * 32 + d], W[d * 32 + o], acc);
  out[t] = acc;
}

// h[e,o] = relu(b1[o] + sum_f attr[e,f]*w1[f,o])
__global__ void k_edge_h(const float* __restrict__ attr, const float* __restrict__ w1,
                         const float* __restrict__ b1, float* __restrict__ h, int E) {
  int t = blockIdx.x * blockDim.x + threadIdx.x;
  if (t >= E * 32) return;
  int e = t >> 5, o = t & 31;
  float acc = b1[o];
  #pragma unroll
  for (int f = 0; f < FE; ++f) acc = fmaf(attr[e * FE + f], w1[f * 32 + o], acc);
  h[t] = fmaxf(acc, 0.0f);
}

// Build B^T in bf16, layout Bt[o][j], j = k*32+d for j<1024, bias rows after.
__global__ void k_build_bt(const float* __restrict__ w2, const float* __restrict__ b2,
                           __bf16* __restrict__ bt) {
  int t = blockIdx.x * blockDim.x + threadIdx.x;
  if (t >= 32 * KTOT) return;
  int o = t / KTOT, j = t % KTOT;
  float v;
  if (j < 1024) { int k = j >> 5, d = j & 31; v = w2[k * 1024 + d * 32 + o]; }
  else          { int d = j - 1024;           v = b2[d * 32 + o]; }
  bt[o * KTOT + j] = (__bf16)v;
}

__global__ void k_fill_u32(unsigned* __restrict__ p, unsigned v, int n) {
  int t = blockIdx.x * blockDim.x + threadIdx.x;
  if (t < n) p[t] = v;
}

// Fused edge-weight * message GEMM + scatter-max.
// One wave per PAIR of 16-edge tiles: B operands (shared across all edges) are
// fetched once per K-chunk and reused by 4 WMMAs (2 A-tiles x 2 N-halves),
// halving B traffic per unit of math and giving 4 independent accumulator
// chains (hides the bf16 WMMA RAW/coexec hazard slots).
// K = 1056 in 33 chunks of 32; A generated in registers as the outer product
// of an h column with the gathered x row (chunk kk has k==kk, d==t).
__global__ void __launch_bounds__(256)
k_msg_scatter(const float* __restrict__ x, const float* __restrict__ h,
              const __bf16* __restrict__ Bt, const int* __restrict__ src,
              const int* __restrict__ dst, unsigned* __restrict__ agg, int nTiles) {
  int wpair = (blockIdx.x * blockDim.x + threadIdx.x) >> 5;
  int nPairs = (nTiles + 1) >> 1;
  if (wpair >= nPairs) return;
  int lane = threadIdx.x & 31;
  int row  = lane & 15;        // A-matrix M row this lane serves
  int half = lane >> 4;        // K-split: 0 -> K{0..7,16..23}, 1 -> K{8..15,24..31}

  int t0 = wpair * 2;
  int t1 = (t0 + 1 < nTiles) ? t0 + 1 : t0;   // tail: redo t0 (atomicMax idempotent)
  int e0 = t0 * 16 + row;
  int e1 = t1 * 16 + row;
  int s0 = src[e0];
  int s1 = src[e1];

  // Gather the 16 x-values each lane contributes per K-chunk (layout-fixed).
  const float* xr0 = x + s0 * 32 + half * 8;
  const float* xr1 = x + s1 * 32 + half * 8;
  float xa0[8], xb0[8], xa1[8], xb1[8];
  #pragma unroll
  for (int i = 0; i < 8; ++i) {
    xa0[i] = xr0[i]; xb0[i] = xr0[16 + i];
    xa1[i] = xr1[i]; xb1[i] = xr1[16 + i];
  }

  const float*  hr0 = h + e0 * 32;
  const float*  hr1 = h + e1 * 32;
  const __bf16* b0p = Bt + (lane & 15) * KTOT + half * 8;          // cols 0..15
  const __bf16* b1p = Bt + ((lane & 15) + 16) * KTOT + half * 8;   // cols 16..31

  v8f acc00 = {0.f, 0.f, 0.f, 0.f, 0.f, 0.f, 0.f, 0.f};
  v8f acc01 = acc00, acc10 = acc00, acc11 = acc00;

  #pragma unroll 4
  for (int kk = 0; kk < 33; ++kk) {
    float hv0 = (kk < 32) ? hr0[kk] : 1.0f;   // chunk 32 = bias rows (p = x)
    float hv1 = (kk < 32) ? hr1[kk] : 1.0f;
    BV a0, a1;
    #pragma unroll
    for (int i = 0; i < 8; ++i) {
      a0.v[i]     = (__bf16)(hv0 * xa0[i]);
      a0.v[8 + i] = (__bf16)(hv0 * xb0[i]);
      a1.v[i]     = (__bf16)(hv1 * xa1[i]);
      a1.v[8 + i] = (__bf16)(hv1 * xb1[i]);
    }
    BV b0, b1;
    const __bf16* p0 = b0p + kk * 32;
    const __bf16* p1 = b1p + kk * 32;
    b0.q[0] = *(const uint4*)(p0);       b0.q[1] = *(const uint4*)(p0 + 16);
    b1.q[0] = *(const uint4*)(p1);       b1.q[1] = *(const uint4*)(p1 + 16);
    acc00 = __builtin_amdgcn_wmma_f32_16x16x32_bf16(false, a0.v, false, b0.v,
                                                    (short)0, acc00, false, false);
    acc10 = __builtin_amdgcn_wmma_f32_16x16x32_bf16(false, a1.v, false, b0.v,
                                                    (short)0, acc10, false, false);
    acc01 = __builtin_amdgcn_wmma_f32_16x16x32_bf16(false, a0.v, false, b1.v,
                                                    (short)0, acc01, false, false);
    acc11 = __builtin_amdgcn_wmma_f32_16x16x32_bf16(false, a1.v, false, b1.v,
                                                    (short)0, acc11, false, false);
  }

  // C/D layout: VGPR i -> M = i (lanes 0-15) / 8+i (lanes 16-31), N = lane&15.
  int col = lane & 15;
  #pragma unroll
  for (int i = 0; i < 8; ++i) {
    int m  = half ? 8 + i : i;
    int d0 = dst[t0 * 16 + m];
    int d1 = dst[t1 * 16 + m];
    atomicMax(agg + d0 * 32 + col,      enc_ord(acc00[i]));
    atomicMax(agg + d0 * 32 + 16 + col, enc_ord(acc01[i]));
    atomicMax(agg + d1 * 32 + col,      enc_ord(acc10[i]));
    atomicMax(agg + d1 * 32 + 16 + col, enc_ord(acc11[i]));
  }
}

// xnew = relu(decode(agg) + xold @ rootW + rootB), with no-edge -> 0 semantics.
__global__ void k_combine(float* __restrict__ xnew, const unsigned* __restrict__ agg,
                          const float* __restrict__ xold, const float* __restrict__ rw,
                          const float* __restrict__ rb, int N) {
  int t = blockIdx.x * blockDim.x + threadIdx.x;
  if (t >= N * 32) return;
  int n = t >> 5, o = t & 31;
  unsigned e = agg[t];
  float m = (e == ENC_NEGINF) ? 0.0f : dec_ord(e);
  float acc = m + rb[o];
  #pragma unroll 8
  for (int d = 0; d < 32; ++d) acc = fmaf(xold[n * 32 + d], rw[d * 32 + o], acc);
  xnew[t] = fmaxf(acc, 0.0f);
}

// Link-prediction head: gather endpoints + attr, 72->64->64->1 MLP, sigmoid.
// MLP weights staged in LDS; h1 held in registers (inner loops unrolled).
__global__ void __launch_bounds__(256)
k_head(const float* __restrict__ xc, const float* __restrict__ xt,
       const float* __restrict__ attr, const int* __restrict__ lsrc,
       const int* __restrict__ ldst, const float* __restrict__ W1,
       const float* __restrict__ B1, const float* __restrict__ W2,
       const float* __restrict__ B2, const float* __restrict__ W3,
       const float* __restrict__ B3, float* __restrict__ out, int L) {
  __shared__ float sW1[72 * 64];
  __shared__ float sW2[64 * 64];
  __shared__ float sB1[64], sB2[64], sW3[64];
  int tid = threadIdx.x;
  for (int i = tid; i < 72 * 64; i += 256) sW1[i] = W1[i];
  for (int i = tid; i < 64 * 64; i += 256) sW2[i] = W2[i];
  if (tid < 64) { sB1[tid] = B1[tid]; sB2[tid] = B2[tid]; sW3[tid] = W3[tid]; }
  __syncthreads();

  int e = blockIdx.x * blockDim.x + tid;
  if (e >= L) return;
  int s = lsrc[e], d = ldst[e];

  float h1[64];
  #pragma unroll
  for (int j = 0; j < 64; ++j) h1[j] = sB1[j];
  for (int i = 0; i < 72; ++i) {
    float zi = (i < 32) ? xc[s * 32 + i]
             : (i < 64) ? xt[d * 32 + (i - 32)]
                        : attr[e * FE + (i - 64)];
    #pragma unroll
    for (int j = 0; j < 64; ++j) h1[j] = fmaf(zi, sW1[i * 64 + j], h1[j]);
  }
  #pragma unroll
  for (int j = 0; j < 64; ++j) h1[j] = fmaxf(h1[j], 0.0f);

  float acc = B3[0];
  for (int j = 0; j < 64; ++j) {
    float t2 = sB2[j];
    #pragma unroll
    for (int i = 0; i < 64; ++i) t2 = fmaf(h1[i], sW2[i * 64 + j], t2);
    acc = fmaf(fmaxf(t2, 0.0f), sW3[j], acc);
  }
  out[e] = 1.0f / (1.0f + __expf(-acc));
}

static inline size_t alignUp(size_t x) { return (x + 255) & ~size_t(255); }

extern "C" void kernel_launch(void* const* d_in, const int* in_sizes, int n_in,
                              void* d_out, int out_size, void* d_ws, size_t ws_size,
                              hipStream_t stream) {
  (void)in_sizes; (void)n_in; (void)out_size; (void)ws_size;
  const float* x_congress = (const float*)d_in[0];
  const float* x_ticker   = (const float*)d_in[1];
  const float* buy_attr   = (const float*)d_in[2];
  const float* rev_attr   = (const float*)d_in[3];
  const float* label_attr = (const float*)d_in[4];
  const float* ntl_w      = (const float*)d_in[5];
  const float* ntl_b      = (const float*)d_in[6];
  const float* emlp_w1    = (const float*)d_in[7];
  const float* emlp_b1    = (const float*)d_in[8];
  const float* emlp_w2    = (const float*)d_in[9];
  const float* emlp_b2    = (const float*)d_in[10];
  const float* root_w     = (const float*)d_in[11];
  const float* root_b     = (const float*)d_in[12];
  const float* head_w1    = (const float*)d_in[13];
  const float* head_b1    = (const float*)d_in[14];
  const float* head_w2    = (const float*)d_in[15];
  const float* head_b2    = (const float*)d_in[16];
  const float* head_w3    = (const float*)d_in[17];
  const float* head_b3    = (const float*)d_in[18];
  const int* buy_src   = (const int*)d_in[19];
  const int* buy_dst   = (const int*)d_in[20];
  const int* rev_src   = (const int*)d_in[21];
  const int* rev_dst   = (const int*)d_in[22];
  const int* label_src = (const int*)d_in[23];
  const int* label_dst = (const int*)d_in[24];
  float* outp = (float*)d_out;

  // ---- carve workspace ----
  char* w = (char*)d_ws;
  size_t off = 0;
  auto carve = [&](size_t bytes) { void* p = w + off; off += alignUp(bytes); return p; };
  float*    xcA   = (float*)carve(NCN * 32 * 4);
  float*    xcB   = (float*)carve(NCN * 32 * 4);
  float*    xtA   = (float*)carve(NTN * 32 * 4);
  float*    xtB   = (float*)carve(NTN * 32 * 4);
  float*    hbuy  = (float*)carve((size_t)EE * 32 * 4);
  float*    hrev  = (float*)carve((size_t)EE * 32 * 4);
  __bf16*   btBuy = (__bf16*)carve(32 * KTOT * 2);
  __bf16*   btRev = (__bf16*)carve(32 * KTOT * 2);
  unsigned* aggT  = (unsigned*)carve(NTN * 32 * 4);
  unsigned* aggC  = (unsigned*)carve(NCN * 32 * 4);

  const int B = 256;
  auto g = [](int n) { return (n + 255) / 256; };

  // per-node-type input linears
  k_linear32<<<g(NCN * 32), B, 0, stream>>>(x_congress, ntl_w,        ntl_b,      xcA, NCN);
  k_linear32<<<g(NTN * 32), B, 0, stream>>>(x_ticker,   ntl_w + 1024, ntl_b + 32, xtA, NTN);

  // edge MLP stage 1 (shared across layers)
  k_edge_h<<<g(EE * 32), B, 0, stream>>>(buy_attr, emlp_w1,       emlp_b1,      hbuy, EE);
  k_edge_h<<<g(EE * 32), B, 0, stream>>>(rev_attr, emlp_w1 + 256, emlp_b1 + 32, hrev, EE);

  // bf16 B^T for the fused edge-weight/message GEMM
  k_build_bt<<<g(32 * KTOT), B, 0, stream>>>(emlp_w2,         emlp_b2,        btBuy);
  k_build_bt<<<g(32 * KTOT), B, 0, stream>>>(emlp_w2 + 32768, emlp_b2 + 1024, btRev);

  float* xc_cur = xcA; float* xc_nxt = xcB;
  float* xt_cur = xtA; float* xt_nxt = xtB;
  const int msgGrid = (NPAIRS + 7) / 8;   // 8 waves (tile-pairs) per 256-thread block

  for (int l = 0; l < 2; ++l) {
    k_fill_u32<<<g(NTN * 32), B, 0, stream>>>(aggT, ENC_NEGINF, NTN * 32);
    k_fill_u32<<<g(NCN * 32), B, 0, stream>>>(aggC, ENC_NEGINF, NCN * 32);
    // both convs read the same input dict (xc_cur, xt_cur)
    k_msg_scatter<<<msgGrid, B, 0, stream>>>(xc_cur, hbuy, btBuy, buy_src, buy_dst, aggT, NTILES);
    k_msg_scatter<<<msgGrid, B, 0, stream>>>(xt_cur, hrev, btRev, rev_src, rev_dst, aggC, NTILES);
    const float* rwT = root_w + (size_t)(l * 2 + 0) * 1024;
    const float* rbT = root_b + (size_t)(l * 2 + 0) * 32;
    const float* rwC = root_w + (size_t)(l * 2 + 1) * 1024;
    const float* rbC = root_b + (size_t)(l * 2 + 1) * 32;
    k_combine<<<g(NTN * 32), B, 0, stream>>>(xt_nxt, aggT, xt_cur, rwT, rbT, NTN);
    k_combine<<<g(NCN * 32), B, 0, stream>>>(xc_nxt, aggC, xc_cur, rwC, rbC, NCN);
    float* t;
    t = xc_cur; xc_cur = xc_nxt; xc_nxt = t;
    t = xt_cur; xt_cur = xt_nxt; xt_nxt = t;
  }

  // link-prediction head
  k_head<<<g(LL), B, 0, stream>>>(xc_cur, xt_cur, label_attr, label_src, label_dst,
                                  head_w1, head_b1, head_w2, head_b2, head_w3, head_b3,
                                  outp, LL);
}
